// MultiHeadAxialAttentionLayer_1872605741674
// MI455X (gfx1250) — matrix operand order repaired and checked by reading, not compiled, not run on hardware
//
#include <hip/hip_runtime.h>

#define EPS 1e-7f

#ifndef __has_builtin
#define __has_builtin(x) 0
#endif
#if __has_builtin(__builtin_amdgcn_tensor_load_to_lds)
#define USE_TDM 1
#else
#define USE_TDM 0
#endif

typedef __attribute__((ext_vector_type(16))) __bf16 v16bf;
typedef __attribute__((ext_vector_type(8)))  __bf16 v8bf;
typedef __attribute__((ext_vector_type(8)))  float  v8f;

__device__ __forceinline__ v8f wmma_bf16(v16bf a, v16bf b, v8f c) {
  return __builtin_amdgcn_wmma_f32_16x16x32_bf16(false, a, false, b, (short)0, c,
                                                 false, false);
}

// A-operand fragment (16x32 bf16). Caller passes pointer to the start of this
// lane's matrix row (row = m0 + (lane&15)). Per ISA layout, lane-half `hseg`
// holds K = [k0+hseg*8, +8) and [k0+16+hseg*8, +8): two b128 loads.
__device__ __forceinline__ v16bf frag_a(const __bf16* rowp, int k0, int hseg) {
  const __bf16* p = rowp + k0 + hseg * 8;
  v8bf x0 = *(const v8bf*)(p);
  v8bf x1 = *(const v8bf*)(p + 16);
  v16bf a;
#pragma unroll
  for (int i = 0; i < 8; ++i) { a[i] = x0[i]; a[i + 8] = x1[i]; }
  return a;
}

// B-operand fragment (32x16 bf16). Source tile stored transposed (row index =
// output column N), so lane-half `hseg` reads K = [k0+hseg*8, +8) and [k0+16+hseg*8, +8).
__device__ __forceinline__ v16bf frag_b(const __bf16* colp, int k0, int hseg) {
  const __bf16* p = colp + k0 + hseg * 8;
  v8bf x0 = *(const v8bf*)(p);
  v8bf x1 = *(const v8bf*)(p + 16);
  v16bf b;
#pragma unroll
  for (int i = 0; i < 8; ++i) { b[i] = x0[i]; b[i + 8] = x1[i]; }
  return b;
}

// fp32 -> bf16 hi + bf16 lo (error-corrected split)
__device__ __forceinline__ void split2(float f, __bf16& h, __bf16& l) {
  h = (__bf16)f;
  l = (__bf16)(f - (float)h);
}

#if USE_TDM
typedef unsigned int v4u_t __attribute__((ext_vector_type(4)));
typedef int v8i_t __attribute__((ext_vector_type(8)));
typedef int v4i_t __attribute__((ext_vector_type(4)));

// Async TDM load of a 128-row x 64-elem bf16 tile (row stride `gstride_elems`)
// into LDS at byte offset `lds_off`. pad_enable inserts 4 DWORDs (16 B) after
// every 32 DWORDs (128 B = one row), producing LDS row stride 144 B = 72 bf16,
// exactly the compute-side fragment layout. Issue from ONE wave only.
__device__ __forceinline__ void tdm_load_k_tile(const __bf16* gsrc,
                                                unsigned lds_off,
                                                unsigned gstride_elems) {
  unsigned long long ga = (unsigned long long)gsrc;
  v4u_t g0;
  g0[0] = 1u;                                   // count=1, user descriptor
  g0[1] = lds_off;                              // LDS byte address
  g0[2] = (unsigned)ga;                         // global_addr[31:0]
  g0[3] = (unsigned)((ga >> 32) & 0x01ffffffu)  // global_addr[56:32]
          | 0x80000000u;                        // type=2 ("image")
  v8i_t g1;
  g1[0] = (int)((1u << 16)      // data_size = 2 bytes
                | (1u << 20)    // pad_enable
                | (4u << 22)    // pad_interval: every 32 DWORDs
                | (3u << 25));  // pad_amount: 4 DWORDs
  g1[1] = (int)(64u << 16);     // tensor_dim0 = 64 (bits 63:48)
  g1[2] = (int)(128u << 16);    // tensor_dim1 = 128 (bits 95:80)
  g1[3] = (int)(64u << 16);     // tile_dim0 = 64 (bits 127:112)
  g1[4] = (int)128u;            // tile_dim1 = 128 (bits 143:128)
  g1[5] = (int)gstride_elems;   // tensor_dim0_stride (bits 191:160)
  g1[6] = 0;
  g1[7] = 0;
  v4i_t z4 = {0, 0, 0, 0};
#if __clang_major__ >= 23
  v8i_t z8 = {0, 0, 0, 0, 0, 0, 0, 0};
  __builtin_amdgcn_tensor_load_to_lds(g0, g1, z4, z4, z8, 0);
#else
  __builtin_amdgcn_tensor_load_to_lds(g0, g1, z4, z4, 0);
#endif
}
#endif  // USE_TDM

// ---------------------------------------------------------------------------
// Kernel 1: fused 1x1-conv projections Q/K/V = X*W + b, stored as bf16 hi/lo
// planes. Grid: (row-tiles=512, col-tiles=8, matrix=3). 256 thr = 8 waves,
// each wave computes a 16x64 strip of a 128x64 tile.
// ---------------------------------------------------------------------------
__global__ __launch_bounds__(256) void qkv_proj_kernel(
    const float* __restrict__ X,
    const float* __restrict__ wq, const float* __restrict__ bq,
    const float* __restrict__ wk, const float* __restrict__ bk,
    const float* __restrict__ wv, const float* __restrict__ bv,
    __bf16* __restrict__ Qh, __bf16* __restrict__ Ql,
    __bf16* __restrict__ Kh, __bf16* __restrict__ Kl,
    __bf16* __restrict__ Vh, __bf16* __restrict__ Vl)
{
  const int row0 = blockIdx.x * 128;
  const int f0   = blockIdx.y * 64;
  const int zm   = blockIdx.z;
  const float* Wm = (zm == 0) ? wq : (zm == 1) ? wk : wv;
  const float* Bm = (zm == 0) ? bq : (zm == 1) ? bk : bv;
  __bf16* Oh = (zm == 0) ? Qh : (zm == 1) ? Kh : Vh;
  __bf16* Ol = (zm == 0) ? Ql : (zm == 1) ? Kl : Vl;

  const int tid  = threadIdx.x;
  const int wv32 = tid >> 5;
  const int lane = tid & 31;
  const int hseg = lane >> 4;
  const int l16  = lane & 15;
  const int m0   = wv32 * 16;

  extern __shared__ char smem[];
  __bf16* sWh = (__bf16*)smem;      // [64][40] transposed weight tile, hi
  __bf16* sWl = sWh + 64 * 40;      // lo

  v8f acc[4] = {};

  const float* xrow = X + (size_t)(row0 + m0 + l16) * 512;
  const int wn  = tid & 63;         // output column within tile (coalesced)
  const int wk0 = (tid >> 6) * 8;   // K sub-chunk

  for (int k0 = 0; k0 < 512; k0 += 32) {
    __syncthreads();
#pragma unroll
    for (int i = 0; i < 8; ++i) {
      float wval = Wm[(size_t)(k0 + wk0 + i) * 512 + f0 + wn];
      split2(wval, sWh[wn * 40 + wk0 + i], sWl[wn * 40 + wk0 + i]);
    }
    __syncthreads();

    // A fragment straight from global X (no cross-wave reuse -> no staging)
    float xr[16];
    {
      const float4 a0 = *(const float4*)(xrow + k0 + hseg * 8);
      const float4 a1 = *(const float4*)(xrow + k0 + hseg * 8 + 4);
      const float4 a2 = *(const float4*)(xrow + k0 + 16 + hseg * 8);
      const float4 a3 = *(const float4*)(xrow + k0 + 16 + hseg * 8 + 4);
      xr[0]=a0.x; xr[1]=a0.y; xr[2]=a0.z;  xr[3]=a0.w;
      xr[4]=a1.x; xr[5]=a1.y; xr[6]=a1.z;  xr[7]=a1.w;
      xr[8]=a2.x; xr[9]=a2.y; xr[10]=a2.z; xr[11]=a2.w;
      xr[12]=a3.x; xr[13]=a3.y; xr[14]=a3.z; xr[15]=a3.w;
    }
    v16bf aH, aL;
#pragma unroll
    for (int i = 0; i < 16; ++i) {
      __bf16 hh, ll; split2(xr[i], hh, ll);
      aH[i] = hh; aL[i] = ll;
    }
#pragma unroll
    for (int t = 0; t < 4; ++t) {
      v16bf bH = frag_b(sWh + (t * 16 + l16) * 40, 0, hseg);
      v16bf bL = frag_b(sWl + (t * 16 + l16) * 40, 0, hseg);
      acc[t] = wmma_bf16(aL, bH, acc[t]);
      acc[t] = wmma_bf16(aH, bL, acc[t]);
      acc[t] = wmma_bf16(aH, bH, acc[t]);
    }
  }

  // epilogue: bias, split to bf16 hi/lo planes
#pragma unroll
  for (int t = 0; t < 4; ++t) {
    const int col = f0 + t * 16 + l16;
    const float bias = Bm[col];
#pragma unroll
    for (int r = 0; r < 8; ++r) {
      const int row = row0 + m0 + r + hseg * 8;
      const float v = acc[t][r] + bias;
      const size_t idx = (size_t)row * 512 + col;
      __bf16 hh, ll; split2(v, hh, ll);
      Oh[idx] = hh; Ol[idx] = ll;
    }
  }
}

// ---------------------------------------------------------------------------
// Kernel 2: column (vertical) attention, fused per (b, head, w).
//   S = clip(Qc Kc^T)/8 -> softmax -> clip ; xv = S @ Vc
// K tile arrives via TDM (hardware-padded to the 72-bf16 row stride); V tile
// is transposed+staged manually. S overlays the K region after GEMM1.
// ---------------------------------------------------------------------------
__global__ __launch_bounds__(256) void axial_col_attn_kernel(
    const __bf16* __restrict__ Qh, const __bf16* __restrict__ Ql,
    const __bf16* __restrict__ Kh, const __bf16* __restrict__ Kl,
    const __bf16* __restrict__ Vh, const __bf16* __restrict__ Vl,
    float* __restrict__ xvb)
{
  const int w = blockIdx.x, n = blockIdx.y, b = blockIdx.z;
  const int tid  = threadIdx.x;
  const int wv32 = tid >> 5, lane = tid & 31, hseg = lane >> 4, l16 = lane & 15;
  const int m0 = wv32 * 16;

  extern __shared__ char smem[];
  __bf16* sKh = (__bf16*)smem;                  // [128][72]
  __bf16* sKl = sKh + 128 * 72;
  __bf16* sSh = (__bf16*)smem;                  // overlay: [128][136]
  __bf16* sSl = sSh + 128 * 136;
  __bf16* sVh = (__bf16*)smem + 2 * 128 * 136;  // [64][136] transposed
  __bf16* sVl = sVh + 64 * 136;

  // K tile: rows g along H, row stride 128*512 elements
  const size_t ktile = ((size_t)(b * 16384 + w)) * 512 + (size_t)n * 64;
#if USE_TDM
  if (tid < 32) {
    tdm_load_k_tile(Kh + ktile, (unsigned)(unsigned long long)(void*)sKh, 65536u);
    tdm_load_k_tile(Kl + ktile, (unsigned)(unsigned long long)(void*)sKl, 65536u);
    __builtin_amdgcn_s_wait_tensorcnt(0);
  }
#endif
  {  // cooperative load: V transposed (+ K fallback without TDM)
    const int g = tid >> 1;
    const int dd0 = (tid & 1) * 32;
    const size_t base = ktile + (size_t)g * 65536 + dd0;
#pragma unroll
    for (int i = 0; i < 4; ++i) {
#if !USE_TDM
      *(v8bf*)(sKh + g * 72 + dd0 + i * 8) = *(const v8bf*)(Kh + base + i * 8);
      *(v8bf*)(sKl + g * 72 + dd0 + i * 8) = *(const v8bf*)(Kl + base + i * 8);
#endif
      v8bf vh = *(const v8bf*)(Vh + base + i * 8);
      v8bf vl = *(const v8bf*)(Vl + base + i * 8);
#pragma unroll
      for (int j = 0; j < 8; ++j) {
        sVh[(dd0 + i * 8 + j) * 136 + g] = vh[j];
        sVl[(dd0 + i * 8 + j) * 136 + g] = vl[j];
      }
    }
  }
  __syncthreads();

  // GEMM1: scores S[m0..m0+15][0..127], A = Q rows from global
  const size_t qoff =
      ((size_t)((b * 128 + (m0 + l16)) * 128 + w)) * 512 + n * 64;
  v8f acc[8] = {};
#pragma unroll
  for (int kt = 0; kt < 2; ++kt) {
    const int k0 = kt * 32;
    v16bf aH = frag_a(Qh + qoff, k0, hseg);
    v16bf aL = frag_a(Ql + qoff, k0, hseg);
#pragma unroll
    for (int t = 0; t < 8; ++t) {
      v16bf bH = frag_b(sKh + (t * 16 + l16) * 72, k0, hseg);
      v16bf bL = frag_b(sKl + (t * 16 + l16) * 72, k0, hseg);
      acc[t] = wmma_bf16(aL, bH, acc[t]);
      acc[t] = wmma_bf16(aH, bL, acc[t]);
      acc[t] = wmma_bf16(aH, bH, acc[t]);
    }
  }

  // clip -> /sqrt(d) -> softmax -> clip. Row lives in one 16-lane half.
#pragma unroll
  for (int r = 0; r < 8; ++r) {
    float v[8];
#pragma unroll
    for (int t = 0; t < 8; ++t)
      v[t] = fminf(fmaxf(acc[t][r], -1.f + EPS), 1.f - EPS) * 0.125f;
    float mx = v[0];
#pragma unroll
    for (int t = 1; t < 8; ++t) mx = fmaxf(mx, v[t]);
#pragma unroll
    for (int d2 = 1; d2 < 16; d2 <<= 1) mx = fmaxf(mx, __shfl_xor(mx, d2, 16));
    float s = 0.f;
#pragma unroll
    for (int t = 0; t < 8; ++t) { v[t] = __expf(v[t] - mx); s += v[t]; }
#pragma unroll
    for (int d2 = 1; d2 < 16; d2 <<= 1) s += __shfl_xor(s, d2, 16);
    const float inv = 1.f / s;
#pragma unroll
    for (int t = 0; t < 8; ++t)
      acc[t][r] = fminf(fmaxf(v[t] * inv, EPS), 1.f - EPS);
  }

  __syncthreads();  // everyone done reading sK before overlay
#pragma unroll
  for (int t = 0; t < 8; ++t) {
#pragma unroll
    for (int r = 0; r < 8; ++r) {
      const int row = m0 + r + hseg * 8;
      const int col = t * 16 + l16;
      split2(acc[t][r], sSh[row * 136 + col], sSl[row * 136 + col]);
    }
  }
  __syncthreads();

  // GEMM2: xv = S @ V
  v8f o[4] = {};
#pragma unroll
  for (int kt = 0; kt < 4; ++kt) {
    const int k0 = kt * 32;
    v16bf aH = frag_a(sSh + (m0 + l16) * 136, k0, hseg);
    v16bf aL = frag_a(sSl + (m0 + l16) * 136, k0, hseg);
#pragma unroll
    for (int t = 0; t < 4; ++t) {
      v16bf bH = frag_b(sVh + (t * 16 + l16) * 136, k0, hseg);
      v16bf bL = frag_b(sVl + (t * 16 + l16) * 136, k0, hseg);
      o[t] = wmma_bf16(aL, bH, o[t]);
      o[t] = wmma_bf16(aH, bL, o[t]);
      o[t] = wmma_bf16(aH, bH, o[t]);
    }
  }

#pragma unroll
  for (int t = 0; t < 4; ++t) {
#pragma unroll
    for (int r = 0; r < 8; ++r) {
      const int hrow = m0 + r + hseg * 8;
      const int dd = t * 16 + l16;
      xvb[(((size_t)(b * 8 + n) * 128 + hrow) * 128 + w) * 64 + dd] = o[t][r];
    }
  }
}

// ---------------------------------------------------------------------------
// Kernel 3: row (horizontal) attention, fused per (b, head, h).
//   S2 = clip(Qr Kr^T)/8 -> softmax -> clip ; out = S2 @ xv_row
// Final store applies the reference channel interleave: ch = d*8 + head.
// ---------------------------------------------------------------------------
__global__ __launch_bounds__(256) void axial_row_attn_kernel(
    const __bf16* __restrict__ Qh, const __bf16* __restrict__ Ql,
    const __bf16* __restrict__ Kh, const __bf16* __restrict__ Kl,
    const float* __restrict__ xvb,
    float* __restrict__ out)
{
  const int hq = blockIdx.x, n = blockIdx.y, b = blockIdx.z;
  const int tid  = threadIdx.x;
  const int wv32 = tid >> 5, lane = tid & 31, hseg = lane >> 4, l16 = lane & 15;
  const int m0 = wv32 * 16;

  extern __shared__ char smem[];
  __bf16* sKh = (__bf16*)smem;                  // [128][72]
  __bf16* sKl = sKh + 128 * 72;
  __bf16* sSh = (__bf16*)smem;                  // overlay: [128][136]
  __bf16* sSl = sSh + 128 * 136;
  __bf16* sXh = (__bf16*)smem + 2 * 128 * 136;  // [64][136] transposed xv
  __bf16* sXl = sXh + 64 * 136;

  // K tile: rows wr along W, row stride 512 elements
  const size_t ktile = ((size_t)((b * 128 + hq) * 128)) * 512 + (size_t)n * 64;
#if USE_TDM
  if (tid < 32) {
    tdm_load_k_tile(Kh + ktile, (unsigned)(unsigned long long)(void*)sKh, 512u);
    tdm_load_k_tile(Kl + ktile, (unsigned)(unsigned long long)(void*)sKl, 512u);
    __builtin_amdgcn_s_wait_tensorcnt(0);
  }
#endif
  {  // cooperative load: fp32 xv tile transposed/split (+ K fallback)
    const int wr = tid >> 1;
    const int dd0 = (tid & 1) * 32;
#if !USE_TDM
    const size_t base = ktile + (size_t)wr * 512 + dd0;
#pragma unroll
    for (int i = 0; i < 4; ++i) {
      *(v8bf*)(sKh + wr * 72 + dd0 + i * 8) = *(const v8bf*)(Kh + base + i * 8);
      *(v8bf*)(sKl + wr * 72 + dd0 + i * 8) = *(const v8bf*)(Kl + base + i * 8);
    }
#endif
    const float* xp =
        xvb + (((size_t)(b * 8 + n) * 128 + hq) * 128 + wr) * 64 + dd0;
#pragma unroll
    for (int i = 0; i < 8; ++i) {
      const float4 f = *(const float4*)(xp + i * 4);
      float fv[4] = {f.x, f.y, f.z, f.w};
#pragma unroll
      for (int c = 0; c < 4; ++c) {
        const int dd = dd0 + i * 4 + c;
        split2(fv[c], sXh[dd * 136 + wr], sXl[dd * 136 + wr]);
      }
    }
  }
  __syncthreads();

  // GEMM1: scores over W
  const size_t qoff =
      ((size_t)((b * 128 + hq) * 128 + (m0 + l16))) * 512 + n * 64;
  v8f acc[8] = {};
#pragma unroll
  for (int kt = 0; kt < 2; ++kt) {
    const int k0 = kt * 32;
    v16bf aH = frag_a(Qh + qoff, k0, hseg);
    v16bf aL = frag_a(Ql + qoff, k0, hseg);
#pragma unroll
    for (int t = 0; t < 8; ++t) {
      v16bf bH = frag_b(sKh + (t * 16 + l16) * 72, k0, hseg);
      v16bf bL = frag_b(sKl + (t * 16 + l16) * 72, k0, hseg);
      acc[t] = wmma_bf16(aL, bH, acc[t]);
      acc[t] = wmma_bf16(aH, bL, acc[t]);
      acc[t] = wmma_bf16(aH, bH, acc[t]);
    }
  }

#pragma unroll
  for (int r = 0; r < 8; ++r) {
    float v[8];
#pragma unroll
    for (int t = 0; t < 8; ++t)
      v[t] = fminf(fmaxf(acc[t][r], -1.f + EPS), 1.f - EPS) * 0.125f;
    float mx = v[0];
#pragma unroll
    for (int t = 1; t < 8; ++t) mx = fmaxf(mx, v[t]);
#pragma unroll
    for (int d2 = 1; d2 < 16; d2 <<= 1) mx = fmaxf(mx, __shfl_xor(mx, d2, 16));
    float s = 0.f;
#pragma unroll
    for (int t = 0; t < 8; ++t) { v[t] = __expf(v[t] - mx); s += v[t]; }
#pragma unroll
    for (int d2 = 1; d2 < 16; d2 <<= 1) s += __shfl_xor(s, d2, 16);
    const float inv = 1.f / s;
#pragma unroll
    for (int t = 0; t < 8; ++t)
      acc[t][r] = fminf(fmaxf(v[t] * inv, EPS), 1.f - EPS);
  }

  __syncthreads();
#pragma unroll
  for (int t = 0; t < 8; ++t) {
#pragma unroll
    for (int r = 0; r < 8; ++r) {
      const int row = m0 + r + hseg * 8;
      const int col = t * 16 + l16;
      split2(acc[t][r], sSh[row * 136 + col], sSl[row * 136 + col]);
    }
  }
  __syncthreads();

  v8f o[4] = {};
#pragma unroll
  for (int kt = 0; kt < 4; ++kt) {
    const int k0 = kt * 32;
    v16bf aH = frag_a(sSh + (m0 + l16) * 136, k0, hseg);
    v16bf aL = frag_a(sSl + (m0 + l16) * 136, k0, hseg);
#pragma unroll
    for (int t = 0; t < 4; ++t) {
      v16bf bH = frag_b(sXh + (t * 16 + l16) * 136, k0, hseg);
      v16bf bL = frag_b(sXl + (t * 16 + l16) * 136, k0, hseg);
      o[t] = wmma_bf16(aL, bH, o[t]);
      o[t] = wmma_bf16(aH, bL, o[t]);
      o[t] = wmma_bf16(aH, bH, o[t]);
    }
  }

#pragma unroll
  for (int t = 0; t < 4; ++t) {
#pragma unroll
    for (int r = 0; r < 8; ++r) {
      const int wrow = m0 + r + hseg * 8;
      const int dd = t * 16 + l16;
      out[((size_t)(b * 128 + hq) * 128 + wrow) * 512 + dd * 8 + n] = o[t][r];
    }
  }
}

// ---------------------------------------------------------------------------
extern "C" void kernel_launch(void* const* d_in, const int* in_sizes, int n_in,
                              void* d_out, int out_size, void* d_ws, size_t ws_size,
                              hipStream_t stream)
{
  const float* x  = (const float*)d_in[0];
  const float* wq = (const float*)d_in[1];
  const float* bq = (const float*)d_in[2];
  const float* wk = (const float*)d_in[3];
  const float* bk = (const float*)d_in[4];
  const float* wv = (const float*)d_in[5];
  const float* bv = (const float*)d_in[6];
  float* out = (float*)d_out;

  const size_t PLANE = (size_t)4 * 128 * 128 * 512;  // 33,554,432 elems
  __bf16* Qh = (__bf16*)d_ws;
  __bf16* Ql = Qh + PLANE;
  __bf16* Kh = Ql + PLANE;
  __bf16* Kl = Kh + PLANE;
  __bf16* Vh = Kl + PLANE;
  __bf16* Vl = Vh + PLANE;
  float* xvbuf = (float*)(Vl + PLANE);  // [4][8][128][128][64] fp32

  qkv_proj_kernel<<<dim3(512, 8, 3), 256, 64 * 40 * 2 * 2, stream>>>(
      x, wq, bq, wk, bk, wv, bv, Qh, Ql, Kh, Kl, Vh, Vl);

  const size_t attn_lds = (2u * 128 * 136 + 2u * 64 * 136) * 2;  // 104448 B
  axial_col_attn_kernel<<<dim3(128, 8, 4), 256, attn_lds, stream>>>(
      Qh, Ql, Kh, Kl, Vh, Vl, xvbuf);
  axial_row_attn_kernel<<<dim3(128, 8, 4), 256, attn_lds, stream>>>(
      Qh, Ql, Kh, Kl, xvbuf, out);

  (void)in_sizes; (void)n_in; (void)out_size; (void)ws_size;
}
